// GIN_66194035966457
// MI455X (gfx1250) — matrix operand order repaired and checked by reading, not compiled, not run on hardware
//
#include <hip/hip_runtime.h>

#define DIM 64
#define LN_EPS 1e-5f

typedef __attribute__((ext_vector_type(16))) __bf16 v16bf;
typedef __attribute__((ext_vector_type(8)))  __bf16 v8bf;
typedef __attribute__((ext_vector_type(8)))  float  v8f;

// hardware float->bf16 conversion path (v_cvt_pk_bf16_f32 class)
static __device__ inline unsigned pack_bf2(float lo, float hi) {
  unsigned short a = __builtin_bit_cast(unsigned short, (__bf16)lo);
  unsigned short b = __builtin_bit_cast(unsigned short, (__bf16)hi);
  return (unsigned)a | ((unsigned)b << 16);
}

// ---------------------------------------------------------------------------
// buf = src (vectorized copy, float4)
// ---------------------------------------------------------------------------
__global__ void copy_f4_kernel(const float* __restrict__ src,
                               float* __restrict__ dst, int n4) {
  int i = blockIdx.x * blockDim.x + threadIdx.x;
  if (i < n4) ((float4*)dst)[i] = ((const float4*)src)[i];
}

// ---------------------------------------------------------------------------
// acc[dst] += x[src] over edges. 16 threads per edge, float4 gather,
// 4 global_atomic_add_f32 (non-returning) per thread. L2-resident target.
// ---------------------------------------------------------------------------
__global__ void scatter_add_kernel(const float* __restrict__ x,
                                   const long long* __restrict__ ei,
                                   float* __restrict__ acc, int E) {
  int tid = blockIdx.x * blockDim.x + threadIdx.x;
  if (tid >= E * 16) return;
  int e = tid >> 4;
  int f = (tid & 15) * 4;
  int src = (int)ei[e];
  int dst = (int)ei[E + e];
  const float4 v = *(const float4*)&x[src * DIM + f];
  float* p = &acc[dst * DIM + f];
  atomicAdd(p + 0, v.x);
  atomicAdd(p + 1, v.y);
  atomicAdd(p + 2, v.z);
  atomicAdd(p + 3, v.w);
}

// ---------------------------------------------------------------------------
// Fused GIN MLP: out = [relu]( LN( LN( in@W1 + b1 )@W2 + b2 ) )
// 256 threads = 8 waves; each wave owns a 16-row tile -> 128 rows per block.
// A-fragments come from a bf16 shadow tile (pure ds_load_b128, no cvt);
// LayerNorm regenerates the bf16 tile for the next layer.
// ---------------------------------------------------------------------------
__global__ __launch_bounds__(256) void gin_mlp_kernel(
    const float* __restrict__ in, float* __restrict__ out,
    const float* __restrict__ w1, const float* __restrict__ b1,
    const float* __restrict__ g1, const float* __restrict__ be1,
    const float* __restrict__ w2, const float* __restrict__ b2,
    const float* __restrict__ g2, const float* __restrict__ be2,
    int N, int applyRelu) {
  __shared__ __align__(32) __bf16 Wt[DIM][DIM];      // transposed: Wt[n][k]
  __shared__ float P1b[DIM], P1g[DIM], P1be[DIM];
  __shared__ float P2b[DIM], P2g[DIM], P2be[DIM];
  __shared__ __align__(16) float  tile [8][16][68];  // fp32 pre-LN scratch
  __shared__ __align__(16) __bf16 tileB[8][16][72];  // bf16 A-source tile

  const int tid  = threadIdx.x;
  const int wid  = tid >> 5;
  const int lane = tid & 31;
  const int half = lane >> 4;
  const int l16  = lane & 15;

  // cooperative: W1 transposed into LDS as bf16 + LN params
  for (int i = tid; i < DIM * DIM; i += 256)
    Wt[i & 63][i >> 6] = (__bf16)w1[i];
  for (int i = tid; i < DIM; i += 256) {
    P1b[i] = b1[i]; P1g[i] = g1[i]; P1be[i] = be1[i];
    P2b[i] = b2[i]; P2g[i] = g2[i]; P2be[i] = be2[i];
  }

  const int rowBase = blockIdx.x * 128 + wid * 16;

  // load 16x64 fp32 rows, convert once to bf16 tile (clamp keeps EXEC full)
  {
    int r  = lane >> 1;
    int c0 = (lane & 1) * 32;
    int grow = rowBase + r;
    if (grow >= N) grow = N - 1;
    const float4* s4 = (const float4*)&in[grow * DIM + c0];
    unsigned* dB = (unsigned*)&tileB[wid][r][c0];
#pragma unroll
    for (int j = 0; j < 8; ++j) {
      float4 v = s4[j];
      dB[2 * j + 0] = pack_bf2(v.x, v.y);
      dB[2 * j + 1] = pack_bf2(v.z, v.w);
    }
  }
  __syncthreads();

  auto do_layer = [&](const float* bb, const float* gg, const float* bbe,
                      bool doRelu, bool writeF32) {
    // A fragments: lane row = l16; for k-block kb the 16 elements are
    // contiguous: [kb*32+half*16, +16)
    v16bf a0, a1;
#pragma unroll
    for (int kb = 0; kb < 2; ++kb) {
      v8bf lo = *(const v8bf*)&tileB[wid][l16][kb * 32 + half * 16];
      v8bf hi = *(const v8bf*)&tileB[wid][l16][kb * 32 + half * 16 + 8];
      v16bf a = __builtin_shufflevector(lo, hi, 0, 1, 2, 3, 4, 5, 6, 7,
                                        8, 9, 10, 11, 12, 13, 14, 15);
      if (kb == 0) a0 = a; else a1 = a;
    }
    // B fragments: lane col = l16; K = kb*32 + half*16 + e (16 contiguous)
    v8f acc[4];
#pragma unroll
    for (int t = 0; t < 4; ++t) {
      const v16bf bk0 = *(const v16bf*)&Wt[t * 16 + l16][half * 16];
      const v16bf bk1 = *(const v16bf*)&Wt[t * 16 + l16][32 + half * 16];
      v8f c = {};
      c = __builtin_amdgcn_wmma_f32_16x16x32_bf16(false, a0, false, bk0,
                                                  (short)0, c, false, false);
      c = __builtin_amdgcn_wmma_f32_16x16x32_bf16(false, a1, false, bk1,
                                                  (short)0, c, false, false);
      acc[t] = c;
    }
    __syncthreads();  // Wt/tiles fully consumed block-wide
    // C layout: vgpr r -> row r + 8*half, col = t*16 + l16 ; add bias
#pragma unroll
    for (int t = 0; t < 4; ++t) {
      int col = t * 16 + l16;
#pragma unroll
      for (int r = 0; r < 8; ++r)
        tile[wid][r + 8 * half][col] = acc[t][r] + bb[col];
    }
    __syncthreads();
    // row-wise LayerNorm: 2 lanes/row, 32 cols each, shuffle-combine;
    // write bf16 tile (next layer's A) and optionally fp32 (final store)
    {
      int r  = lane >> 1;
      int c0 = (lane & 1) * 32;
      float v[32];
      float s = 0.f, sq = 0.f;
#pragma unroll
      for (int j = 0; j < 32; ++j) {
        v[j] = tile[wid][r][c0 + j];
        s += v[j]; sq += v[j] * v[j];
      }
      s  += __shfl_xor(s, 1, 32);
      sq += __shfl_xor(sq, 1, 32);
      float mean = s * (1.f / 64.f);
      float var  = sq * (1.f / 64.f) - mean * mean;
      float rs   = rsqrtf(var + LN_EPS);
      unsigned* dB = (unsigned*)&tileB[wid][r][c0];
#pragma unroll
      for (int j = 0; j < 32; ++j) {
        int c = c0 + j;
        float o = (v[j] - mean) * rs * gg[c] + bbe[c];
        if (doRelu) o = fmaxf(o, 0.f);
        v[j] = o;
        if (writeF32) tile[wid][r][c] = o;
      }
#pragma unroll
      for (int j = 0; j < 16; ++j)
        dB[j] = pack_bf2(v[2 * j], v[2 * j + 1]);
    }
    __syncthreads();
  };

  do_layer(P1b, P1g, P1be, false, false);

  // swap weights: Wt <- W2^T (brackets are uniform __syncthreads)
  for (int i = tid; i < DIM * DIM; i += 256)
    Wt[i & 63][i >> 6] = (__bf16)w2[i];
  __syncthreads();

  do_layer(P2b, P2g, P2be, applyRelu != 0, true);

  // store result rows from fp32 tile
  {
    int r  = lane >> 1;
    int c0 = (lane & 1) * 32;
    int grow = rowBase + r;
    if (grow < N) {
      float4* d4 = (float4*)&out[grow * DIM + c0];
      const float4* s4 = (const float4*)&tile[wid][r][c0];
#pragma unroll
      for (int j = 0; j < 8; ++j) d4[j] = s4[j];
    }
  }
}

// ---------------------------------------------------------------------------
// segment-sum pooling accumulators (atomics; G*D = 32 KB, L2-resident)
// ---------------------------------------------------------------------------
__global__ void pool_accum_kernel(const float* __restrict__ h,
                                  const long long* __restrict__ batch,
                                  float* __restrict__ sums,
                                  float* __restrict__ counts, int N) {
  int tid = blockIdx.x * blockDim.x + threadIdx.x;
  if (tid >= N * DIM) return;
  int n = tid >> 6;
  int d = tid & 63;
  int g = (int)batch[n];
  atomicAdd(&sums[g * DIM + d], h[tid]);
  if (d == 0) atomicAdd(&counts[g], 1.0f);
}

// ---------------------------------------------------------------------------
// out[g] = (sums[g]/max(count,1)) @ wl + bl   (tiny: G=128 rows)
// ---------------------------------------------------------------------------
__global__ void final_linear_kernel(const float* __restrict__ sums,
                                    const float* __restrict__ counts,
                                    const float* __restrict__ wl,
                                    const float* __restrict__ bl,
                                    float* __restrict__ out) {
  __shared__ float row[DIM];
  int g = blockIdx.x;
  int d = threadIdx.x;
  float c = fmaxf(counts[g], 1.0f);
  row[d] = sums[g * DIM + d] / c;
  __syncthreads();
  float acc = bl[d];
#pragma unroll 8
  for (int k = 0; k < DIM; ++k) acc += row[k] * wl[k * DIM + d];
  out[g * DIM + d] = acc;
}

// ---------------------------------------------------------------------------
extern "C" void kernel_launch(void* const* d_in, const int* in_sizes, int n_in,
                              void* d_out, int out_size, void* d_ws,
                              size_t ws_size, hipStream_t stream) {
  const float*     x     = (const float*)d_in[0];
  const long long* ei    = (const long long*)d_in[1];
  const long long* batch = (const long long*)d_in[2];
  const float *w11 = (const float*)d_in[3],  *b11 = (const float*)d_in[4];
  const float *g11 = (const float*)d_in[5],  *be11 = (const float*)d_in[6];
  const float *w12 = (const float*)d_in[7],  *b12 = (const float*)d_in[8];
  const float *g12 = (const float*)d_in[9],  *be12 = (const float*)d_in[10];
  const float *w21 = (const float*)d_in[11], *b21 = (const float*)d_in[12];
  const float *g21 = (const float*)d_in[13], *be21 = (const float*)d_in[14];
  const float *w22 = (const float*)d_in[15], *b22 = (const float*)d_in[16];
  const float *g22 = (const float*)d_in[17], *be22 = (const float*)d_in[18];
  const float *wl  = (const float*)d_in[19], *bl  = (const float*)d_in[20];
  float* out = (float*)d_out;

  const int N = in_sizes[0] / DIM;
  const int E = in_sizes[1] / 2;
  const int G = out_size / DIM;

  float* bufA   = (float*)d_ws;            // N*DIM : conv input (x + agg)
  float* bufB   = bufA + (size_t)N * DIM;  // N*DIM : conv output
  float* sums   = bufB + (size_t)N * DIM;  // G*DIM
  float* counts = sums + (size_t)G * DIM;  // G

  const int n4 = N * DIM / 4;
  const int copyBlocks = (n4 + 255) / 256;
  const int scatBlocks = (E * 16 + 255) / 256;
  const int mlpBlocks  = (N + 127) / 128;
  const int poolBlocks = (N * DIM + 255) / 256;

  // ---- conv1: bufA = x + segment_sum(x[src] -> dst); bufB = MLP(bufA) ----
  copy_f4_kernel<<<copyBlocks, 256, 0, stream>>>(x, bufA, n4);
  scatter_add_kernel<<<scatBlocks, 256, 0, stream>>>(x, ei, bufA, E);
  gin_mlp_kernel<<<mlpBlocks, 256, 0, stream>>>(
      bufA, bufB, w11, b11, g11, be11, w12, b12, g12, be12, N, 1);

  // ---- conv2 ----
  copy_f4_kernel<<<copyBlocks, 256, 0, stream>>>(bufB, bufA, n4);
  scatter_add_kernel<<<scatBlocks, 256, 0, stream>>>(bufB, ei, bufA, E);
  gin_mlp_kernel<<<mlpBlocks, 256, 0, stream>>>(
      bufA, bufB, w21, b21, g21, be21, w22, b22, g22, be22, N, 1);

  // ---- global mean pool + final linear ----
  hipMemsetAsync(sums, 0, (size_t)(G * DIM + G) * sizeof(float), stream);
  pool_accum_kernel<<<poolBlocks, 256, 0, stream>>>(bufB, batch, sums, counts, N);
  final_linear_kernel<<<G, DIM, 0, stream>>>(sums, counts, wl, bl, out);
}
